// NeRFModel_22222160789633
// MI455X (gfx1250) — matrix operand order repaired and checked by reading, not compiled, not run on hardware
//
#include <hip/hip_runtime.h>
#include <hip/hip_bf16.h>

// ---------------------------------------------------------------------------
// NeRF forward for MI455X (gfx1250).
// bf16 WMMA GEMM chain for the MLP. A-tiles staged to LDS with
// global_load_async_to_lds_b128 (ASYNCcnt) and shared by all 8 waves;
// each wave computes a 16x64 output tile (4 accumulators) so one A
// fragment feeds 4 v_wmma_f32_16x16x32_bf16. Weights pre-packed to the
// WMMA B-fragment layout (~1.2MB, L2-resident).
// ---------------------------------------------------------------------------

typedef unsigned short u16;
typedef unsigned int   u32;

#define NUM_COARSE 64
#define NUM_FINE   128
#define BATCH      1024
#define WIDTH      256
#define LDA        320            // activation row stride in halves
#define M_COARSE   (BATCH * NUM_COARSE)   // 65536
#define M_FINE     (BATCH * NUM_FINE)     // 131072
#define LASTD      0.0001f
#define EPSR       1e-05f
#define PI_F       3.14159265358979323846f

__device__ __forceinline__ u16 f2bf(float f) {
    u32 u = __float_as_uint(f);
    u32 r = (u >> 16) & 1u;
    u += 0x7fffu + r;              // round-to-nearest-even
    return (u16)(u >> 16);
}
__device__ __forceinline__ float bf2f(u16 h) {
    return __uint_as_float(((u32)h) << 16);
}

// ---------------------------------------------------------------------------
// WMMA types
// ---------------------------------------------------------------------------
typedef __attribute__((ext_vector_type(16))) __bf16 v16bf;
typedef __attribute__((ext_vector_type(8)))  float  v8f;

union FragU { uint4 u[2]; v16bf v; };

// ---------------------------------------------------------------------------
// Weight packing: W is fp32 [Korig x N] row-major (fan_in x fan_out).
// Packed per (kt, nt) 32x16 tile: lane l stores 16 contiguous bf16:
//   lane <16 : column n = nt*16+l,    K = kt*32 + 0..15
//   lane >=16: column n = nt*16+l-16, K = kt*32 + 16..31
// Fragment stride = 512 halves. K zero-padded Korig -> Kp.
// ---------------------------------------------------------------------------
__global__ void pack_w_kernel(const float* __restrict__ W, u16* __restrict__ out,
                              int Korig, int N, int Kp)
{
    int tid = blockIdx.x * blockDim.x + threadIdx.x;
    int nTiles = N >> 4;
    int frags = (Kp >> 5) * nTiles;
    if (tid >= frags * 32) return;
    int lane = tid & 31;
    int frag = tid >> 5;
    int kt = frag / nTiles;
    int nt = frag % nTiles;
    int n  = nt * 16 + (lane & 15);
    int k0 = kt * 32 + ((lane >> 4) ? 16 : 0);
    u16* dst = out + (size_t)frag * 512 + (size_t)lane * 16;
    #pragma unroll
    for (int j = 0; j < 16; ++j) {
        int k = k0 + j;
        float v = (k < Korig) ? W[(size_t)k * N + n] : 0.0f;
        dst[j] = f2bf(v);
    }
}

// ---------------------------------------------------------------------------
// WMMA GEMM: C[M x N] = act(A[M x Kp] * B + bias)
// Block = 256 threads = 8 wave32. nSubs = N/64 waves along N, 8/nSubs along
// M; block tile = (8/nSubs*16) rows x N cols. The block's A tile
// (rows x Kp bf16) is async-copied to LDS once (ASYNCcnt), then every wave
// reads A fragments from LDS and B fragments from the packed (L2-resident)
// weights; one A fragment is reused by 4 WMMAs (16x64 per wave). All 4 B
// fragments are fetched before the WMMA quad so their loads clause together
// and the waits are progressive instead of serializing each WMMA.
// ---------------------------------------------------------------------------
__global__ __launch_bounds__(256) void wmma_gemm_kernel(
    const u16* __restrict__ A, int lda,
    const u16* __restrict__ Bp,
    const float* __restrict__ bias,
    u16* __restrict__ C, int ldc,
    int Kp, int Ntot, int relu)
{
    extern __shared__ u16 tileA[];          // rowsPerBlock x Kp halves

    const int lane   = threadIdx.x & 31;
    const int wave   = threadIdx.x >> 5;
    const int nSubs  = Ntot >> 6;           // waves along N (64 cols each)
    const int wavesM = 8 / nSubs;
    const int rowsPerBlock = wavesM << 4;
    const int mBase  = blockIdx.x * rowsPerBlock;
    const int mSub   = wave / nSubs;
    const int nSub   = wave % nSubs;
    const int col    = lane & 15;
    const int hi     = lane >> 4;           // lane group 0/1
    const int nTiles = Ntot >> 4;

    // ---- stage A block tile into LDS via async copy (16B per lane/op) ----
    {
        const int chunksPerRow = Kp >> 3;   // 16-byte chunks per row
        const int total = rowsPerBlock * chunksPerRow;
        for (int c = threadIdx.x; c < total; c += 256) {
            int row = c / chunksPerRow;
            int off = (c - row * chunksPerRow) << 3;          // halves
            const u16* g = A + (size_t)(mBase + row) * lda + off;
            u32 ldsOff = (u32)(uintptr_t)(tileA + row * Kp + off);
            asm volatile("global_load_async_to_lds_b128 %0, %1, off"
                         :: "v"(ldsOff), "v"(g) : "memory");
        }
    }

    // ---- accumulators preloaded with bias (uniform per column) ----
    v8f acc[4];
    #pragma unroll
    for (int j = 0; j < 4; ++j) {
        float b = bias[nSub * 64 + j * 16 + col];
        v8f t = {b, b, b, b, b, b, b, b};
        acc[j] = t;
    }

    asm volatile("s_wait_asynccnt 0x0" ::: "memory");
    __syncthreads();

    // ---- K loop: A fragment from LDS, 4 B fragments prefetched, 4 WMMAs ----
    const u16* aLane = tileA + (size_t)(mSub * 16 + col) * Kp + hi * 16;
    const int kSteps = Kp >> 5;
    for (int kt = 0; kt < kSteps; ++kt) {
        FragU a;
        const u16* ap = aLane + kt * 32;
        a.u[0] = *(const uint4*)(ap);        // K 0-7   (or 16-23)
        a.u[1] = *(const uint4*)(ap + 8);    // K 8-15  (or 24-31)

        FragU b[4];
        #pragma unroll
        for (int j = 0; j < 4; ++j) {
            const u16* bp = Bp + (((size_t)kt * nTiles + (nSub * 4 + j)) * 32 + lane) * 16;
            b[j].u[0] = *(const uint4*)(bp);
            b[j].u[1] = *(const uint4*)(bp + 8);
        }
        #pragma unroll
        for (int j = 0; j < 4; ++j) {
            acc[j] = __builtin_amdgcn_wmma_f32_16x16x32_bf16(
                         false, a.v, false, b[j].v, (short)0, acc[j], false, false);
        }
    }

    // ---- store: C/D layout VGPR i -> row +i (+8 for hi lanes) ----
    #pragma unroll
    for (int j = 0; j < 4; ++j) {
        u16* crow = C + (size_t)(mBase + mSub * 16 + hi * 8) * ldc
                      + nSub * 64 + j * 16 + col;
        #pragma unroll
        for (int i = 0; i < 8; ++i) {
            float v = acc[j][i];
            if (relu) v = fmaxf(v, 0.0f);
            crow[(size_t)i * ldc] = f2bf(v);
        }
    }
}

// ---------------------------------------------------------------------------
// t_coarse generation: t = near + (far-near) * n/63
// ---------------------------------------------------------------------------
__global__ void tgen_coarse_kernel(const float* __restrict__ nearp,
                                   const float* __restrict__ farp,
                                   float* __restrict__ tco)
{
    int i = blockIdx.x * blockDim.x + threadIdx.x;
    if (i >= BATCH * NUM_COARSE) return;
    int b = i >> 6, n = i & 63;
    tco[i] = nearp[b] + (farp[b] - nearp[b]) * ((float)n * (1.0f / 63.0f));
}

// ---------------------------------------------------------------------------
// Ray setup + positional encoding.
// Writes: penc   [r*64 + 0..59] bf16, 60..63 zero   (layer-0 A)
//         Q      [r*320 + 256..315] = p_enc, 316..319 zero (skip concat)
//         denc   [r*24 + 0..23] bf16
// ---------------------------------------------------------------------------
__global__ void encode_kernel(const float* __restrict__ hor,
                              const float* __restrict__ ver,
                              const float* __restrict__ tm,
                              const float* __restrict__ kinv,
                              const float* __restrict__ tarr,
                              int nsShift, int Mrows,
                              u16* __restrict__ penc,
                              u16* __restrict__ Q,
                              u16* __restrict__ denc)
{
    int r = blockIdx.x * blockDim.x + threadIdx.x;
    if (r >= Mrows) return;
    int b = r >> nsShift;
    float t = tarr[r];
    float x = hor[b], y = ver[b];
    float ps0 = kinv[0] * x + kinv[1] * y + kinv[2];
    float ps1 = kinv[3] * x + kinv[4] * y + kinv[5];
    float ps2 = kinv[6] * x + kinv[7] * y + kinv[8];
    float pc0 = ps0 * t, pc1 = ps1 * t, pc2 = ps2 * t;
    float nrm = fmaxf(sqrtf(ps0 * ps0 + ps1 * ps1 + ps2 * ps2), 1e-12f);
    float d0 = ps0 / nrm, d1 = ps1 / nrm, d2 = ps2 / nrm;
    const float* M = tm + (size_t)b * 16;
    float pw[3], dw[3];
    #pragma unroll
    for (int i = 0; i < 3; ++i) {
        pw[i] = M[i * 4 + 0] * pc0 + M[i * 4 + 1] * pc1 + M[i * 4 + 2] * pc2 + M[i * 4 + 3];
        dw[i] = M[i * 4 + 0] * d0  + M[i * 4 + 1] * d1  + M[i * 4 + 2] * d2;
    }
    // p_enc: [dim][freq][sin,cos], freq_l = 2^(10*l/9) * pi
    u16* pe = penc + (size_t)r * 64;
    u16* qe = Q + (size_t)r * LDA + 256;
    #pragma unroll
    for (int d = 0; d < 3; ++d) {
        #pragma unroll
        for (int l = 0; l < 10; ++l) {
            float f = exp2f((10.0f / 9.0f) * (float)l) * PI_F;
            float g = pw[d] * f;
            u16 s = f2bf(__sinf(g));
            u16 cc = f2bf(__cosf(g));
            pe[d * 20 + 2 * l]     = s;  qe[d * 20 + 2 * l]     = s;
            pe[d * 20 + 2 * l + 1] = cc; qe[d * 20 + 2 * l + 1] = cc;
        }
    }
    #pragma unroll
    for (int j = 60; j < 64; ++j) { pe[j] = 0; qe[j] = 0; }
    // d_enc: freq_l = 2^(4*l/3) * pi
    u16* de = denc + (size_t)r * 24;
    #pragma unroll
    for (int d = 0; d < 3; ++d) {
        #pragma unroll
        for (int l = 0; l < 4; ++l) {
            float f = exp2f((4.0f / 3.0f) * (float)l) * PI_F;
            float g = dw[d] * f;
            de[d * 8 + 2 * l]     = f2bf(__sinf(g));
            de[d * 8 + 2 * l + 1] = f2bf(__cosf(g));
        }
    }
}

// ---------------------------------------------------------------------------
// Copy d_enc into cols 0..23 of P and zero cols 280..287 (dir-layer input:
// [d_enc(24) | pinfo(256) written by GEMM | zeros(8)], K=288).
// ---------------------------------------------------------------------------
__global__ void prep_dir_kernel(const u16* __restrict__ denc,
                                u16* __restrict__ P, int Mrows)
{
    int r = blockIdx.x * blockDim.x + threadIdx.x;
    if (r >= Mrows) return;
    u16* p = P + (size_t)r * LDA;
    const u16* de = denc + (size_t)r * 24;
    #pragma unroll
    for (int j = 0; j < 24; ++j) p[j] = de[j];
    #pragma unroll
    for (int j = 280; j < 288; ++j) p[j] = 0;
}

// ---------------------------------------------------------------------------
// sigma head: |h(256) . sW + sb|
// ---------------------------------------------------------------------------
__global__ void head_sigma_kernel(const u16* __restrict__ H, int ldc,
                                  const float* __restrict__ sW,
                                  const float* __restrict__ sb,
                                  float* __restrict__ sig, int Mrows)
{
    int r = blockIdx.x * blockDim.x + threadIdx.x;
    if (r >= Mrows) return;
    const u16* h = H + (size_t)r * ldc;
    float a = sb[0];
    for (int k = 0; k < WIDTH; ++k) a += bf2f(h[k]) * sW[k];
    sig[r] = fabsf(a);
}

// ---------------------------------------------------------------------------
// color head: sigmoid(cin(128) @ cW(128x3) + cb)
// ---------------------------------------------------------------------------
__global__ void head_color_kernel(const u16* __restrict__ H, int ldc,
                                  const float* __restrict__ cW,
                                  const float* __restrict__ cb,
                                  float* __restrict__ col, int Mrows)
{
    int r = blockIdx.x * blockDim.x + threadIdx.x;
    if (r >= Mrows) return;
    const u16* h = H + (size_t)r * ldc;
    float a0 = cb[0], a1 = cb[1], a2 = cb[2];
    for (int k = 0; k < 128; ++k) {
        float v = bf2f(h[k]);
        a0 += v * cW[k * 3 + 0];
        a1 += v * cW[k * 3 + 1];
        a2 += v * cW[k * 3 + 2];
    }
    col[(size_t)r * 3 + 0] = 1.0f / (1.0f + __expf(-a0));
    col[(size_t)r * 3 + 1] = 1.0f / (1.0f + __expf(-a1));
    col[(size_t)r * 3 + 2] = 1.0f / (1.0f + __expf(-a2));
}

// ---------------------------------------------------------------------------
// Inverse-CDF resampling (one thread per ray).
// ---------------------------------------------------------------------------
__global__ void resample_kernel(const float* __restrict__ tco,
                                const float* __restrict__ sig_co,
                                float* __restrict__ tfi)
{
    int b = blockIdx.x * blockDim.x + threadIdx.x;
    if (b >= BATCH) return;
    float cdf[NUM_COARSE];
    float run = 0.0f;
    for (int i = 0; i < NUM_COARSE; ++i) {
        run += sig_co[b * NUM_COARSE + i];
        cdf[i] = run;
    }
    float lo = cdf[0], hiv = cdf[NUM_COARSE - 1];
    float dg = tco[1] - tco[0];   // reference uses ray-0 spacing globally
    for (int j = 0; j < NUM_FINE; ++j) {
        float u  = (float)(j + 1) * (1.0f / (float)(NUM_FINE + 1));
        float tv = lo + (hiv - lo) * u;
        int sidx = NUM_COARSE;
        for (int i = 0; i < NUM_COARSE; ++i) {
            if (cdf[i] >= tv) { sidx = i; break; }
        }
        int idx = sidx - 1;
        if (idx < 0) idx = 0;
        if (idx > NUM_COARSE - 1) idx = NUM_COARSE - 1;
        float lt    = tco[b * NUM_COARSE + idx];
        float lcdf  = cdf[idx];
        float lslope = (idx < NUM_COARSE - 1)
                           ? dg / (sig_co[b * NUM_COARSE + idx + 1] + EPSR)
                           : 0.0f;
        tfi[b * NUM_FINE + j] = lt + (tv - lcdf) * lslope;
    }
}

// ---------------------------------------------------------------------------
// Final render (one thread per ray): coarse accumulation + merge/sort/fine.
// Reference sorts each of the 5 bundle channels INDEPENDENTLY (jnp.sort
// axis=1 on (B,192,5)) -- we reproduce that.
// ---------------------------------------------------------------------------
__device__ void isort192(float* a) {
    for (int i = 1; i < 192; ++i) {
        float key = a[i];
        int j = i - 1;
        while (j >= 0 && a[j] > key) { a[j + 1] = a[j]; --j; }
        a[j + 1] = key;
    }
}

__global__ void render_kernel(const float* __restrict__ tco,
                              const float* __restrict__ tfi,
                              const float* __restrict__ sig_co,
                              const float* __restrict__ col_co,
                              const float* __restrict__ sig_fi,
                              const float* __restrict__ col_fi,
                              const float* __restrict__ nearp,
                              const float* __restrict__ farp,
                              float* __restrict__ out)
{
    int b = blockIdx.x * blockDim.x + threadIdx.x;
    if (b >= BATCH) return;
    // ---- coarse ----
    float dco = (farp[b] - nearp[b]) * (1.0f / (float)NUM_COARSE);
    float cum = 0.0f, a0 = 0.0f, a1 = 0.0f, a2 = 0.0f;
    for (int i = 0; i < NUM_COARSE; ++i) {
        float sd = dco * sig_co[b * NUM_COARSE + i];
        cum += sd;
        float T = __expf(-cum);
        float w = T * (1.0f - __expf(-sd));
        a0 += col_co[(size_t)(b * NUM_COARSE + i) * 3 + 0] * w;
        a1 += col_co[(size_t)(b * NUM_COARSE + i) * 3 + 1] * w;
        a2 += col_co[(size_t)(b * NUM_COARSE + i) * 3 + 2] * w;
    }
    out[b * 3 + 0] = a0;
    out[b * 3 + 1] = a1;
    out[b * 3 + 2] = a2;
    // ---- fine: merge 64 + 128 ----
    float ts[192], sg[192], c0[192], c1[192], c2[192];
    for (int i = 0; i < NUM_COARSE; ++i) {
        ts[i] = tco[b * NUM_COARSE + i];
        sg[i] = sig_co[b * NUM_COARSE + i];
        c0[i] = col_co[(size_t)(b * NUM_COARSE + i) * 3 + 0];
        c1[i] = col_co[(size_t)(b * NUM_COARSE + i) * 3 + 1];
        c2[i] = col_co[(size_t)(b * NUM_COARSE + i) * 3 + 2];
    }
    for (int i = 0; i < NUM_FINE; ++i) {
        ts[64 + i] = tfi[b * NUM_FINE + i];
        sg[64 + i] = sig_fi[b * NUM_FINE + i];
        c0[64 + i] = col_fi[(size_t)(b * NUM_FINE + i) * 3 + 0];
        c1[64 + i] = col_fi[(size_t)(b * NUM_FINE + i) * 3 + 1];
        c2[64 + i] = col_fi[(size_t)(b * NUM_FINE + i) * 3 + 2];
    }
    isort192(ts); isort192(sg); isort192(c0); isort192(c1); isort192(c2);
    cum = 0.0f; a0 = a1 = a2 = 0.0f;
    for (int i = 0; i < 192; ++i) {
        float dlt = (i < 191) ? (ts[i + 1] - ts[i]) : LASTD;
        float sd = dlt * sg[i];
        cum += sd;
        float T = __expf(-cum);
        float w = T * (1.0f - __expf(-sd));
        a0 += c0[i] * w; a1 += c1[i] * w; a2 += c2[i] * w;
    }
    out[BATCH * 3 + b * 3 + 0] = a0;
    out[BATCH * 3 + b * 3 + 1] = a1;
    out[BATCH * 3 + b * 3 + 2] = a2;
}

// ---------------------------------------------------------------------------
// Host-side launch
// ---------------------------------------------------------------------------
extern "C" void kernel_launch(void* const* d_in, const int* in_sizes, int n_in,
                              void* d_out, int out_size, void* d_ws, size_t ws_size,
                              hipStream_t stream)
{
    (void)in_sizes; (void)n_in; (void)out_size; (void)ws_size;

    // Inputs in setup_inputs() dict (insertion) order; params nested in
    // insertion order: point_W[0..7], point_b[0..7], sigma_W, sigma_b,
    // pinfo_W, pinfo_b, dir_W, dir_b, color_W, color_b.
    const float* hor  = (const float*)d_in[0];
    const float* ver  = (const float*)d_in[1];
    const float* tm   = (const float*)d_in[2];
    const float* kinv = (const float*)d_in[3];
    const float* nrp  = (const float*)d_in[4];
    const float* frp  = (const float*)d_in[5];
    const float* pW[8]; const float* pb[8];
    for (int i = 0; i < 8; ++i) { pW[i] = (const float*)d_in[6 + i]; pb[i] = (const float*)d_in[14 + i]; }
    const float* sW  = (const float*)d_in[22];
    const float* sb  = (const float*)d_in[23];
    const float* piW = (const float*)d_in[24];
    const float* pib = (const float*)d_in[25];
    const float* dW  = (const float*)d_in[26];
    const float* db  = (const float*)d_in[27];
    const float* cW  = (const float*)d_in[28];
    const float* cb  = (const float*)d_in[29];

    // ---- workspace carve ----
    char* base = (char*)d_ws;
    size_t off = 0;
    auto carve = [&](size_t bytes) -> char* {
        char* p = base + off;
        off += (bytes + 255) & ~(size_t)255;
        return p;
    };
    u16* bufP  = (u16*)carve((size_t)M_FINE * LDA * 2);
    u16* bufQ  = (u16*)carve((size_t)M_FINE * LDA * 2);
    u16* penc  = (u16*)carve((size_t)M_FINE * 64 * 2);
    u16* denc  = (u16*)carve((size_t)M_FINE * 24 * 2);
    float* tco    = (float*)carve((size_t)BATCH * NUM_COARSE * 4);
    float* tfi    = (float*)carve((size_t)BATCH * NUM_FINE * 4);
    float* sig_co = (float*)carve((size_t)M_COARSE * 4);
    float* col_co = (float*)carve((size_t)M_COARSE * 3 * 4);
    float* sig_fi = (float*)carve((size_t)M_FINE * 4);
    float* col_fi = (float*)carve((size_t)M_FINE * 3 * 4);
    // packed weights: frag count = (Kp/32)*(N/16), 1024 B per fragment
    auto packBytes = [](int Kp, int N) -> size_t { return (size_t)(Kp / 32) * (N / 16) * 1024; };
    u16* pk[8];
    const int Ko[8] = {60, 256, 256, 256, 316, 256, 256, 256};
    const int Kp[8] = {64, 256, 256, 256, 320, 256, 256, 256};
    for (int i = 0; i < 8; ++i) pk[i] = (u16*)carve(packBytes(Kp[i], 256));
    u16* pkinfo = (u16*)carve(packBytes(256, 256));
    u16* pkdir  = (u16*)carve(packBytes(288, 128));

    const int TB = 256;

    // ---- pack weights (bf16 WMMA B-fragments) ----
    auto pack = [&](const float* W, u16* dst, int Korig, int N, int Kpad) {
        int threads = (Kpad / 32) * (N / 16) * 32;
        pack_w_kernel<<<(threads + TB - 1) / TB, TB, 0, stream>>>(W, dst, Korig, N, Kpad);
    };
    for (int i = 0; i < 8; ++i) pack(pW[i], pk[i], Ko[i], 256, Kp[i]);
    pack(piW, pkinfo, 256, 256, 256);
    pack(dW, pkdir, 280, 128, 288);

    // ---- coarse sample positions ----
    tgen_coarse_kernel<<<(BATCH * NUM_COARSE + TB - 1) / TB, TB, 0, stream>>>(nrp, frp, tco);

    // ---- GEMM launcher ----
    auto gemm = [&](int M, const u16* A, int lda, const u16* Bp, const float* bias,
                    u16* C, int ldc, int Kpad, int N, int relu) {
        int nSubs = N / 64;                 // waves along N
        int rows  = (8 / nSubs) * 16;       // block tile rows
        dim3 grid(M / rows, 1);
        size_t shmem = (size_t)rows * Kpad * 2;
        wmma_gemm_kernel<<<grid, dim3(256), shmem, stream>>>(
            A, lda, Bp, bias, C, ldc, Kpad, N, relu);
    };

    // ---- one full MLP pass (coarse or fine) ----
    auto run_pass = [&](int M, int nsShift, const float* tarr, float* sig, float* col) {
        int gb = (M + TB - 1) / TB;
        encode_kernel<<<gb, TB, 0, stream>>>(hor, ver, tm, kinv, tarr, nsShift, M, penc, bufQ, denc);
        // layer chain (ping-pong P/Q, skip concat lives in Q cols 256..319)
        gemm(M, penc, 64,  pk[0], pb[0], bufP, LDA, 64,  256, 1);   // L0
        gemm(M, bufP, LDA, pk[1], pb[1], bufQ, LDA, 256, 256, 1);   // L1
        gemm(M, bufQ, LDA, pk[2], pb[2], bufP, LDA, 256, 256, 1);   // L2
        gemm(M, bufP, LDA, pk[3], pb[3], bufQ, LDA, 256, 256, 1);   // L3 -> [h3|p_enc] in Q
        gemm(M, bufQ, LDA, pk[4], pb[4], bufP, LDA, 320, 256, 1);   // L4 (skip, K=320)
        gemm(M, bufP, LDA, pk[5], pb[5], bufQ, LDA, 256, 256, 1);   // L5
        gemm(M, bufQ, LDA, pk[6], pb[6], bufP, LDA, 256, 256, 1);   // L6
        gemm(M, bufP, LDA, pk[7], pb[7], bufQ, LDA, 256, 256, 1);   // L7 -> h in Q
        head_sigma_kernel<<<gb, TB, 0, stream>>>(bufQ, LDA, sW, sb, sig, M);
        gemm(M, bufQ, LDA, pkinfo, pib, bufP + 24, LDA, 256, 256, 0); // pinfo -> P cols 24..279
        prep_dir_kernel<<<gb, TB, 0, stream>>>(denc, bufP, M);        // P cols 0..23 & 280..287
        gemm(M, bufP, LDA, pkdir, db, bufQ, LDA, 288, 128, 1);        // dir -> Q cols 0..127
        head_color_kernel<<<gb, TB, 0, stream>>>(bufQ, LDA, cW, cb, col, M);
    };

    // coarse pass
    run_pass(M_COARSE, 6, tco, sig_co, col_co);
    // importance resampling
    resample_kernel<<<(BATCH + TB - 1) / TB, TB, 0, stream>>>(tco, sig_co, tfi);
    // fine pass
    run_pass(M_FINE, 7, tfi, sig_fi, col_fi);
    // final render
    render_kernel<<<(BATCH + TB - 1) / TB, TB, 0, stream>>>(
        tco, tfi, sig_co, col_co, sig_fi, col_fi, nrp, frp, (float*)d_out);
}
